// SCAM_39917426049708
// MI455X (gfx1250) — compile-verified, matrix operand order checked by reading, not run on hardware
//
#include <hip/hip_runtime.h>
#include <hip/hip_bf16.h>

// ---------------- CDNA5 WMMA plumbing (wave32, 16x16x32 bf16 -> f32) --------
typedef __attribute__((ext_vector_type(16))) __bf16 v16bf;
typedef __attribute__((ext_vector_type(8)))  float  v8f;

#define WSZ   16
#define OWSZ  24
#define RTAB  1209          // (2*16-1)*(16+24-1)
#define SCALE 0.25f         // HEAD_DIM^-0.5, HEAD_DIM=16
#define NPIX  32768         // 2*128*128
#define NPE   2097152       // NPIX*64

__device__ __forceinline__ int wmma_k_of(int lane, int i) {
    // 16-bit A/B layout: elements 0..7 <-> K 0..7 (+16 for elems 8..15),
    // lanes 16..31 shifted by +8 in K.
    return (i & 7) + ((i >> 3) << 4) + ((lane >> 4) << 3);
}

__device__ __forceinline__ v8f wmma_bf16(v16bf a, v16bf b, v8f c) {
    return __builtin_amdgcn_wmma_f32_16x16x32_bf16(false, a, false, b,
                                                   (short)0, c, false, false);
}

// A[m][k] row-major (stride), tile at (row0, col0); kmax = valid K within tile
__device__ __forceinline__ v16bf load_a(const __bf16* A, int stride,
                                        int row0, int col0, int kmax) {
    int lane = threadIdx.x & 31;
    int m = row0 + (lane & 15);
    v16bf a;
#pragma unroll
    for (int i = 0; i < 16; ++i) {
        int k = wmma_k_of(lane, i);
        a[i] = (k < kmax) ? A[m * stride + col0 + k] : (__bf16)0.0f;
    }
    return a;
}

// A fragment sourced from an f32 LDS matrix (convert on the fly; co-executes
// with the WMMA/XDL pipe)
__device__ __forceinline__ v16bf load_a_f32(const float* A, int stride,
                                            int row0, int col0) {
    int lane = threadIdx.x & 31;
    int m = row0 + (lane & 15);
    v16bf a;
#pragma unroll
    for (int i = 0; i < 16; ++i) {
        int k = wmma_k_of(lane, i);
        a[i] = (__bf16)A[m * stride + col0 + k];
    }
    return a;
}

// Logical B[k][n] stored row-major (stride over n), tile at (row0=k, col0=n)
__device__ __forceinline__ v16bf load_b(const __bf16* B, int stride,
                                        int row0, int col0, int kmax) {
    int lane = threadIdx.x & 31;
    int n = col0 + (lane & 15);
    v16bf b;
#pragma unroll
    for (int i = 0; i < 16; ++i) {
        int k = wmma_k_of(lane, i);
        b[i] = (k < kmax) ? B[(row0 + k) * stride + n] : (__bf16)0.0f;
    }
    return b;
}

// Logical B[k][n] = M[n][k] (i.e. B = M^T), M row-major with stride over k
__device__ __forceinline__ v16bf load_bt(const __bf16* M, int stride,
                                         int col0 /*n=row of M*/, int krow0, int kmax) {
    int lane = threadIdx.x & 31;
    int n = col0 + (lane & 15);
    v16bf b;
#pragma unroll
    for (int i = 0; i < 16; ++i) {
        int k = wmma_k_of(lane, i);
        b[i] = (k < kmax) ? M[n * stride + krow0 + k] : (__bf16)0.0f;
    }
    return b;
}

// ---------------- Stage A: per-pixel channel LayerNorm, NCHW f32 -> NHWC bf16
__global__ void ln_nhwc_kernel(const float* __restrict__ src,
                               const float* __restrict__ lnw,
                               const float* __restrict__ lnb,
                               int do_ln,
                               __bf16* __restrict__ dst) {
    int pg = blockIdx.x * 256 + threadIdx.x;      // 0..32767
    int b = pg >> 14, rem = pg & 16383;
    const float* base = src + (size_t)b * 64 * 16384 + rem;
    if (do_ln) {
        float s = 0.f, s2 = 0.f;
#pragma unroll 8
        for (int c = 0; c < 64; ++c) { float v = base[c * 16384]; s += v; s2 += v * v; }
        float mu  = s * (1.f / 64.f);
        float inv = rsqrtf(s2 * (1.f / 64.f) - mu * mu + 1e-6f);
#pragma unroll 8
        for (int c = 0; c < 64; ++c)
            dst[(size_t)pg * 64 + c] = (__bf16)((base[c * 16384] - mu) * inv * lnw[c] + lnb[c]);
    } else {
#pragma unroll 8
        for (int c = 0; c < 64; ++c)
            dst[(size_t)pg * 64 + c] = (__bf16)base[c * 16384];
    }
}

// ---------------- Stage B: overlapping cross attention (one window/head/qchunk)
// grid: (qchunk=4, head=4, window=128); block 256 (8 waves)
// LDS: Qs 2K | Ks 12K | scr 4.75K (rpb, then softmax scratch, then reduce) | Sf 96K
__global__ void oca_kernel(const __bf16* __restrict__ q_nhwc,
                           const __bf16* __restrict__ kv_nhwc,
                           const float*  __restrict__ rpb,
                           float* __restrict__ out_nhwc) {
    extern __shared__ char smem[];
    __bf16* Qs  = (__bf16*)smem;                          // 64*16  (2 KB)
    __bf16* Ks  = (__bf16*)(smem + 2048);                 // 384*16 (12 KB)
    float*  scr = (float*)(smem + 2048 + 12288);          // 1216 f32 (4.75 KB)
    float*  Sf  = (float*)(smem + 2048 + 12288 + 4864);   // 64*384 f32 (96 KB)

    int qc = blockIdx.x, hd = blockIdx.y, wi = blockIdx.z;
    int b = wi >> 6, widx = wi & 63, wy = widx >> 3, wx = widx & 7;
    int t = threadIdx.x;

    for (int e = t; e < 64 * 16; e += 256) {              // Q chunk
        int p = e >> 4, d = e & 15;
        int q = qc * 64 + p, qi = q >> 4, qj = q & 15;
        int y = wy * 16 + qi, x = wx * 16 + qj;
        Qs[e] = q_nhwc[(((size_t)(b * 128 + y) * 128 + x) * 64) + hd * 16 + d];
    }
    for (int e = t; e < 384 * 16; e += 256) {             // unfolded K (=V)
        int kk = e >> 4, d = e & 15;
        int ki = kk / 24, kj = kk % 24;
        int y = wy * 16 - 4 + ki, x = wx * 16 - 4 + kj;   // PAD=4 both dims
        __bf16 v = (__bf16)0.0f;
        if (y >= 0 && y < 128 && x >= 0 && x < 128)
            v = kv_nhwc[(((size_t)(b * 128 + y) * 128 + x) * 64) + hd * 16 + d];
        Ks[e] = v;
    }
    for (int e = t; e < RTAB; e += 256) scr[e] = rpb[e * 4 + hd]; // this head's bias
    __syncthreads();

    int wave = t >> 5, lane = t & 31;
    // S = Q K^T * scale + rel-pos bias : 4x24 tiles, 12 per wave
    for (int tt = wave * 12; tt < wave * 12 + 12; ++tt) {
        int mt = tt / 24, nt = tt % 24;
        v16bf a  = load_a (Qs, 16, mt * 16, 0, 16);
        v16bf bb = load_bt(Ks, 16, nt * 16, 0, 16);
        v8f c = {};
        c = wmma_bf16(a, bb, c);
        int n  = nt * 16 + (lane & 15);
        int ki = n / 24, kj = n % 24;
#pragma unroll
        for (int r = 0; r < 8; ++r) {
            int m = mt * 16 + r + ((lane >> 4) << 3);
            int q = qc * 64 + m, qi = q >> 4, qj = q & 15;
            // idx in [-456, 512]: one conditional add IS np.mod (no division)
            int idx = (ki - qi + 1) * 31 + (kj - qj - 7);
            if (idx < 0) idx += RTAB;
            Sf[m * 384 + n] = c[r] * SCALE + scr[idx];
        }
    }
    __syncthreads();
    // -------- parallel softmax: 4 threads per row, 96 cols each ------------
    int row = t >> 2, part = t & 3;
    int c0 = part * 96;
    {
        float mx = -1e30f;
        for (int n = c0; n < c0 + 96; ++n) mx = fmaxf(mx, Sf[row * 384 + n]);
        scr[t] = mx;                                      // scratch reuse (rpb dead)
    }
    __syncthreads();
    float m4 = fmaxf(fmaxf(scr[row * 4 + 0], scr[row * 4 + 1]),
                     fmaxf(scr[row * 4 + 2], scr[row * 4 + 3]));
    __syncthreads();                                      // protect maxes
    {
        float sum = 0.f;
        for (int n = c0; n < c0 + 96; ++n) {
            float e = __expf(Sf[row * 384 + n] - m4);
            Sf[row * 384 + n] = e;
            sum += e;
        }
        scr[t] = sum;
    }
    __syncthreads();
    {
        float s4 = (scr[row * 4 + 0] + scr[row * 4 + 1]) +
                   (scr[row * 4 + 2] + scr[row * 4 + 3]);
        float inv = 1.f / s4;
        for (int n = c0; n < c0 + 96; ++n) Sf[row * 384 + n] *= inv;  // Sf := P (f32)
    }
    __syncthreads();
    // -------- O = P @ K, all 8 waves: split K=384 in halves, LDS-reduce ----
    int mt = wave & 3;
    int ks0 = (wave < 4) ? 0 : 6;
    v8f acc = {};
    for (int ks = ks0; ks < ks0 + 6; ++ks) {
        v16bf a  = load_a_f32(Sf, 384, mt * 16, ks * 32);
        v16bf bb = load_b(Ks, 16, ks * 32, 0, 32);
        acc = wmma_bf16(a, bb, acc);
    }
    if (wave >= 4) {
#pragma unroll
        for (int r = 0; r < 8; ++r) scr[mt * 256 + r * 32 + lane] = acc[r];
    }
    __syncthreads();
    if (wave < 4) {
        int d = lane & 15;
#pragma unroll
        for (int r = 0; r < 8; ++r) {
            float v = acc[r] + scr[mt * 256 + r * 32 + lane];
            int m = mt * 16 + r + ((lane >> 4) << 3);
            int q = qc * 64 + m, qi = q >> 4, qj = q & 15;
            int y = wy * 16 + qi, x = wx * 16 + qj;
            out_nhwc[(((size_t)(b * 128 + y) * 128 + x) * 64) + hd * 16 + d] = v;
        }
    }
}

// ---------------- Stage C: optional LN + 1x1 conv (64x64 GEMM), 128 pixels/block
__global__ void proj_kernel(const float* __restrict__ src, int src_is_nchw,
                            int do_ln,
                            const float* __restrict__ lnw, const float* __restrict__ lnb,
                            const float* __restrict__ W,   const float* __restrict__ bias,
                            __bf16* __restrict__ dst) {
    extern __shared__ char smem[];
    __bf16* As = (__bf16*)smem;                   // 128*64 bf16 (16 KB)
    __bf16* Bs = (__bf16*)(smem + 16384);         // 64*64  bf16 (8 KB)
    float*  Xf = (float*)(smem + 16384 + 8192);   // 128*64 f32  (32 KB)
    int t = threadIdx.x;
    int pbase = blockIdx.x * 128;
    __builtin_prefetch(W + t * 16, 0, 1);         // global_prefetch_b8 on weights

    for (int e = t; e < 128 * 64; e += 256) {
        int p = e >> 6, c = e & 63;
        int pg = pbase + p;
        float v;
        if (src_is_nchw) { int b = pg >> 14, rem = pg & 16383; v = src[(size_t)(b * 64 + c) * 16384 + rem]; }
        else             { v = src[(size_t)pg * 64 + c]; }
        Xf[e] = v;
    }
    for (int e = t; e < 64 * 64; e += 256) {      // B[k=c_in][n=c_out] = W[o][c]
        int o = e >> 6, c = e & 63;
        Bs[c * 64 + o] = (__bf16)W[o * 64 + c];
    }
    __syncthreads();
    if (do_ln) {
        // 2 threads per pixel: thread pair (p, half) each handles 32 channels
        int p = t >> 1, half = t & 1;
        if (p < 128) {
            float s = 0.f, s2 = 0.f;
            for (int c = 0; c < 64; ++c) { float v = Xf[p * 64 + c]; s += v; s2 += v * v; }
            float mu  = s * (1.f / 64.f);
            float inv = rsqrtf(s2 * (1.f / 64.f) - mu * mu + 1e-6f);
            for (int c = half * 32; c < half * 32 + 32; ++c)
                As[p * 64 + c] = (__bf16)((Xf[p * 64 + c] - mu) * inv * lnw[c] + lnb[c]);
        }
    } else {
        for (int e = t; e < 128 * 64; e += 256) As[e] = (__bf16)Xf[e];
    }
    __syncthreads();
    int wave = t >> 5, lane = t & 31;
    for (int tt = wave * 4; tt < wave * 4 + 4; ++tt) {  // 8x4 = 32 tiles
        int mt = tt >> 2, nt = tt & 3;
        v8f acc = {};
        for (int ks = 0; ks < 2; ++ks) {
            v16bf a  = load_a(As, 64, mt * 16, ks * 32, 32);
            v16bf bb = load_b(Bs, 64, ks * 32, nt * 16, 32);
            acc = wmma_bf16(a, bb, acc);
        }
        int o = nt * 16 + (lane & 15);
        float bi = bias[o];
#pragma unroll
        for (int r = 0; r < 8; ++r) {
            int p = mt * 16 + r + ((lane >> 4) << 3);
            dst[(size_t)(pbase + p) * 64 + o] = (__bf16)(acc[r] + bi);
        }
    }
}

// ---------------- Stage D: per-row 128x128 cross attention, dual softmax, residual
// grid: (row=128, b=2); block 256
__global__ void cross_kernel(const __bf16* __restrict__ Ql_, const __bf16* __restrict__ Qr_,
                             const __bf16* __restrict__ Vl_, const __bf16* __restrict__ Vr_,
                             const float* __restrict__ x_l,  const float* __restrict__ x_r,
                             const float* __restrict__ beta, const float* __restrict__ gamma,
                             float* __restrict__ out) {
    extern __shared__ char smem[];
    __bf16* Ql  = (__bf16*)smem;            // each 128*64 bf16 (16 KB)
    __bf16* Qr  = Ql + 128 * 64;
    __bf16* Vl  = Qr + 128 * 64;
    __bf16* Vr  = Vl + 128 * 64;
    float*  Sf  = (float*)(smem + 65536);   // 128*128 f32 (64 KB)
    __bf16* P1  = (__bf16*)(smem + 131072); // 32 KB
    __bf16* P2T = P1 + 128 * 128;           // 32 KB (stored transposed)

    int y = blockIdx.x, b = blockIdx.y;
    int t = threadIdx.x;
    size_t rowoff = ((size_t)(b * 128 + y)) * 128 * 64;
    for (int e = t; e < 128 * 64; e += 256) {
        Ql[e] = Ql_[rowoff + e];
        Qr[e] = Qr_[rowoff + e];
        Vl[e] = Vl_[rowoff + e];
        Vr[e] = Vr_[rowoff + e];
    }
    __syncthreads();
    int wave = t >> 5, lane = t & 31;
    for (int tt = wave * 8; tt < wave * 8 + 8; ++tt) {   // S = Ql Qr^T * scale
        int mt = tt >> 3, nt = tt & 7;
        v8f acc = {};
        for (int ks = 0; ks < 2; ++ks) {
            v16bf a  = load_a (Ql, 64, mt * 16, ks * 32, 32);
            v16bf bb = load_bt(Qr, 64, nt * 16, ks * 32, 32);
            acc = wmma_bf16(a, bb, acc);
        }
        int n = nt * 16 + (lane & 15);
#pragma unroll
        for (int r = 0; r < 8; ++r) {
            int m = mt * 16 + r + ((lane >> 4) << 3);
            Sf[m * 128 + n] = acc[r] * SCALE;
        }
    }
    __syncthreads();
    if (t < 128) {                    // softmax over r (axis=-1) -> P1
        int l = t;
        float mx = -1e30f;
        for (int r = 0; r < 128; ++r) mx = fmaxf(mx, Sf[l * 128 + r]);
        float sum = 0.f;
        for (int r = 0; r < 128; ++r) sum += __expf(Sf[l * 128 + r] - mx);
        float inv = 1.f / sum;
        for (int r = 0; r < 128; ++r) P1[l * 128 + r] = (__bf16)(__expf(Sf[l * 128 + r] - mx) * inv);
    } else {                          // softmax over l (axis=2) -> P2^T
        int r = t - 128;
        float mx = -1e30f;
        for (int l = 0; l < 128; ++l) mx = fmaxf(mx, Sf[l * 128 + r]);
        float sum = 0.f;
        for (int l = 0; l < 128; ++l) sum += __expf(Sf[l * 128 + r] - mx);
        float inv = 1.f / sum;
        for (int l = 0; l < 128; ++l) P2T[r * 128 + l] = (__bf16)(__expf(Sf[l * 128 + r] - mx) * inv);
    }
    __syncthreads();
    // waves 0..3: O1 = P1 @ Vr (-> out_l); waves 4..7: O2 = P2^T @ Vl (-> out_r)
    bool second = wave >= 4;
    const __bf16* Pm = second ? P2T : P1;
    const __bf16* Vm = second ? Vl  : Vr;
    const float*  xin = second ? x_r : x_l;
    const float*  scv = second ? gamma : beta;
    size_t outbase = second ? (size_t)NPE : 0;
    int w4 = wave & 3;
    for (int tt = w4 * 8; tt < w4 * 8 + 8; ++tt) {  // 8x4 = 32 tiles per half
        int mt = tt >> 2, nt = tt & 3;
        v8f acc = {};
        for (int ks = 0; ks < 4; ++ks) {
            v16bf a  = load_a(Pm, 128, mt * 16, ks * 32, 32);
            v16bf bb = load_b(Vm, 64, ks * 32, nt * 16, 32);
            acc = wmma_bf16(a, bb, acc);
        }
        int c = nt * 16 + (lane & 15);
        float scl = scv[c];
#pragma unroll
        for (int r = 0; r < 8; ++r) {
            int m = mt * 16 + r + ((lane >> 4) << 3);
            size_t idx = (((size_t)b * 64 + c) * 128 + y) * 128 + m;
            out[outbase + idx] = xin[idx] + acc[r] * scl;
        }
    }
}

// --------------------------------------------------------------------------
extern "C" void kernel_launch(void* const* d_in, const int* in_sizes, int n_in,
                              void* d_out, int out_size, void* d_ws, size_t ws_size,
                              hipStream_t stream) {
    (void)in_sizes; (void)n_in; (void)out_size; (void)ws_size;
    const float* x_l       = (const float*)d_in[0];
    const float* x_r       = (const float*)d_in[1];
    const float* rpb       = (const float*)d_in[2];
    const float* norm_l_w  = (const float*)d_in[3];
    const float* norm_l_b  = (const float*)d_in[4];
    const float* norm_r_w  = (const float*)d_in[5];
    const float* norm_r_b  = (const float*)d_in[6];
    const float* l_proj1_w = (const float*)d_in[7];
    const float* l_proj1_b = (const float*)d_in[8];
    const float* r_proj1_w = (const float*)d_in[9];
    const float* r_proj1_b = (const float*)d_in[10];
    const float* l_proj2_w = (const float*)d_in[11];
    const float* l_proj2_b = (const float*)d_in[12];
    const float* r_proj2_w = (const float*)d_in[13];
    const float* r_proj2_b = (const float*)d_in[14];
    const float* beta      = (const float*)d_in[15];
    const float* gamma     = (const float*)d_in[16];

    char* ws = (char*)d_ws;
    __bf16* nq_l = (__bf16*)(ws + 0);                 // LN(x_l) NHWC bf16
    __bf16* nq_r = (__bf16*)(ws + 4194304);           // LN(x_r) with norm_l (per ref)
    __bf16* kv_l = (__bf16*)(ws + 8388608);           // raw x_l NHWC bf16
    __bf16* kv_r = (__bf16*)(ws + 12582912);          // raw x_r NHWC bf16
    float*  xl_  = (float*)(ws + 16777216);           // OCA out left, NHWC f32
    float*  xr_  = (float*)(ws + 25165824);           // OCA out right
    __bf16* Qlb  = (__bf16*)(ws + 33554432);
    __bf16* Qrb  = (__bf16*)(ws + 37748736);
    __bf16* Vlb  = (__bf16*)(ws + 41943040);
    __bf16* Vrb  = (__bf16*)(ws + 46137344);

    // Stage A: LN + layout conversion (note: ref normalizes BOTH with norm_l)
    ln_nhwc_kernel<<<128, 256, 0, stream>>>(x_l, norm_l_w, norm_l_b, 1, nq_l);
    ln_nhwc_kernel<<<128, 256, 0, stream>>>(x_r, norm_l_w, norm_l_b, 1, nq_r);
    ln_nhwc_kernel<<<128, 256, 0, stream>>>(x_l, norm_l_w, norm_l_b, 0, kv_l);
    ln_nhwc_kernel<<<128, 256, 0, stream>>>(x_r, norm_l_w, norm_l_b, 0, kv_r);

    // Stage B: overlapping cross attention (K doubles as V)
    size_t oca_lds = 2048 + 12288 + 4864 + 98304;     // ~117 KB -> 2 WGs / WGP
    oca_kernel<<<dim3(4, 4, 128), 256, oca_lds, stream>>>(nq_l, kv_r, rpb, xl_);
    oca_kernel<<<dim3(4, 4, 128), 256, oca_lds, stream>>>(nq_r, kv_l, rpb, xr_);

    // Stage C: LN + 1x1 conv projections (WMMA GEMM)
    size_t proj_lds = 16384 + 8192 + 32768;
    proj_kernel<<<256, 256, proj_lds, stream>>>(xl_, 0, 1, norm_l_w, norm_l_b, l_proj1_w, l_proj1_b, Qlb);
    proj_kernel<<<256, 256, proj_lds, stream>>>(xr_, 0, 1, norm_r_w, norm_r_b, r_proj1_w, r_proj1_b, Qrb);
    proj_kernel<<<256, 256, proj_lds, stream>>>(x_l, 1, 0, norm_l_w, norm_l_b, l_proj2_w, l_proj2_b, Vlb);
    proj_kernel<<<256, 256, proj_lds, stream>>>(x_r, 1, 0, norm_r_w, norm_r_b, r_proj2_w, r_proj2_b, Vrb);

    // Stage D: per-row cross attention + dual softmax + fused residual
    size_t cross_lds = 65536 + 65536 + 32768 + 32768; // 192 KB
    cross_kernel<<<dim3(128, 2), 256, cross_lds, stream>>>(Qlb, Qrb, Vlb, Vrb,
                                                           x_l, x_r, beta, gamma,
                                                           (float*)d_out);
}
